// DistanceAttentionLayer_69535520522570
// MI455X (gfx1250) — matrix operand order repaired and checked by reading, not compiled, not run on hardware
//
#include <hip/hip_runtime.h>
#include <hip/hip_bf16.h>

// ---------------------------------------------------------------------------
// DistanceAttentionLayer for MI455X (gfx1250, wave32, WMMA).
//   B=4, S=2048, D=512, H=8, DK=64, DFF=2048
// All GEMMs via v_wmma_f32_16x16x32_f16 (f16 operands, f32 accum).
// Attention is flash-style: S x S logits never hit HBM.
// GEMM: 128x128 block tile, 8 waves, double-buffered LDS (one barrier/chunk).
// ---------------------------------------------------------------------------

typedef __attribute__((ext_vector_type(16))) _Float16 v16h;
typedef __attribute__((ext_vector_type(8)))  _Float16 v8h;
typedef __attribute__((ext_vector_type(8)))  float    v8f;

#define LANE (threadIdx.x & 31)
#define WID  (threadIdx.x >> 5)

// --- Fragment loaders (layouts per CDNA5 ISA 7.12.2, wave32) ----------------
// A (16x32 f16, MxK), row-major source with `stride` halves per row:
//   lane<16 : row=lane,    halves[0..7]=K 0..7,  halves[8..15]=K 16..23
//   lane>=16: row=lane-16, halves[0..7]=K 8..15, halves[8..15]=K 24..31
__device__ inline v16h load_frag_a(const _Float16* base, int stride) {
  const int lane = LANE;
  const _Float16* p = base + (lane & 15) * stride + ((lane >> 4) << 3);
  v8h lo = *(const v8h*)(p);
  v8h hi = *(const v8h*)(p + 16);
  return __builtin_shufflevector(lo, hi, 0,1,2,3,4,5,6,7,8,9,10,11,12,13,14,15);
}
// B (32x16 f16, KxN) from column-major [n][k] storage (stride halves per col):
//   lane<16 : col=lane,    halves[0..7]=K 0..7,  halves[8..15]=K 16..23
//   lane>=16: col=lane-16, halves[0..7]=K 8..15, halves[8..15]=K 24..31
__device__ inline v16h load_frag_b(const _Float16* base, int stride) {
  const int lane = LANE;
  const _Float16* p = base + (lane & 15) * stride + ((lane >> 4) << 3);
  v8h lo = *(const v8h*)(p);
  v8h hi = *(const v8h*)(p + 16);
  return __builtin_shufflevector(lo, hi, 0,1,2,3,4,5,6,7,8,9,10,11,12,13,14,15);
}

// ---------------------------------------------------------------------------
// Generic WMMA GEMM:  C(M,N) = A(M,K) @ Bw(K,N) [+ bias][+ resid][relu/scale]
// Block tile 128x128 (8 waves x 16 rows x 128 cols), K-chunks of 32,
// double-buffered LDS: stage chunk i+1 under the WMMAs of chunk i.
// MODE 0: out f16 row-major, +bias, optional relu          (FFN1)
// MODE 1: out f16 in (B,H,S,DK) layout, (acc+bias)*scale   (Q/K/V proj)
// MODE 2: out f32 row-major, +bias +resid                  (Wo proj, FFN2)
// M,N,K all multiples of tile sizes for this problem -> no edge guards.
// ---------------------------------------------------------------------------
template<int MODE, typename AT>
__global__ __launch_bounds__(256)
void gemm_wmma_kernel(const AT* __restrict__ A, const float* __restrict__ Bw,
                      const float* __restrict__ bias, const float* __restrict__ resid,
                      void* __restrict__ outp, int M, int N, int K,
                      float scale, int relu)
{
  __shared__ __align__(16) _Float16 ldsA[2][128 * 40];  // [row][k] pad 40
  __shared__ __align__(16) _Float16 ldsB[2][128 * 40];  // [n][k]   pad 40
  const int tid = threadIdx.x;
  const int bm = blockIdx.x * 128, bn = blockIdx.y * 128;

  // stage A chunk (128x32) -> f16 LDS
  auto stageA = [&](int k0, int buf) {
    const int row = tid >> 1, cb = (tid & 1) * 16;
    const AT* src = A + (size_t)(bm + row) * K + k0 + cb;
    _Float16* dst = ldsA[buf] + row * 40 + cb;
#pragma unroll
    for (int j = 0; j < 16; ++j) dst[j] = (_Float16)src[j];
    if (k0 + 32 < K) __builtin_prefetch(src + 32, 0, 0);  // global_prefetch_b8
  };
  // stage B chunk (32x128) transposed -> ldsB[n][k], f16
  auto stageB = [&](int k0, int buf) {
    const int k = tid >> 3, nb = (tid & 7) * 16;
    const float* src = Bw + (size_t)(k0 + k) * N + bn + nb;
#pragma unroll
    for (int j = 0; j < 16; ++j) ldsB[buf][(nb + j) * 40 + k] = (_Float16)src[j];
    if (k0 + 32 < K) __builtin_prefetch(src + (size_t)32 * N, 0, 0);
  };

  v8f acc[8] = {};

  stageA(0, 0); stageB(0, 0);
  __syncthreads();

  int cur = 0;
  for (int k0 = 0; k0 < K; k0 += 32) {
    const int nxt = cur ^ 1;
    if (k0 + 32 < K) { stageA(k0 + 32, nxt); stageB(k0 + 32, nxt); }

    v16h a = load_frag_a(ldsA[cur] + WID * 16 * 40, 40);
#pragma unroll
    for (int t = 0; t < 8; ++t) {
      v16h b = load_frag_b(ldsB[cur] + t * 16 * 40, 40);
      acc[t] = __builtin_amdgcn_wmma_f32_16x16x32_f16(
          false, a, false, b, (short)0, acc[t], false, false);
    }
    __syncthreads();   // next chunk fully staged AND this chunk's reads done
    cur = nxt;
  }

  // Epilogue.  C layout (ISA): VGPR r -> M = r (lanes 0-15) / r+8 (lanes 16-31),
  // N = lane & 15 within each 16-col tile.
  const int lane = LANE;
  const int mbase = bm + WID * 16 + ((lane >> 4) << 3);
  const int nl = lane & 15;
#pragma unroll
  for (int t = 0; t < 8; ++t) {
    const int n = bn + t * 16 + nl;
    const float bv = bias[n];
#pragma unroll
    for (int r = 0; r < 8; ++r) {
      const int m = mbase + r;
      float v = acc[t][r] + bv;
      if constexpr (MODE == 0) {
        if (relu) v = fmaxf(v, 0.f);
        ((_Float16*)outp)[(size_t)m * N + n] = (_Float16)v;
      } else if constexpr (MODE == 1) {
        v *= scale;
        const int b = m >> 11, s = m & 2047, h = n >> 6, dk = n & 63;
        ((_Float16*)outp)[((size_t)(b * 8 + h) * 2048 + s) * 64 + dk] = (_Float16)v;
      } else {
        v += resid[(size_t)m * N + n];
        ((float*)outp)[(size_t)m * N + n] = v;
      }
    }
  }
}

// ---------------------------------------------------------------------------
// Flash attention with multiplicative distance bias + key padding mask.
// Grid: (S/64, B*H); block: 128 (4 waves). Wave w owns q-rows q0..q0+15.
// Q pre-scaled by 1/sqrt(DK) at projection time.
// ---------------------------------------------------------------------------
__global__ __launch_bounds__(128)
void attn_kernel(const _Float16* __restrict__ Qf, const _Float16* __restrict__ Kf,
                 const _Float16* __restrict__ Vf, const float* __restrict__ dist,
                 const unsigned char* __restrict__ mask, _Float16* __restrict__ outp)
{
  __shared__ __align__(16) _Float16 ldsK[32 * 80];    // [key][dk]  pad 80
  __shared__ __align__(16) _Float16 ldsV[64 * 40];    // [dk][key]  pad 40
  __shared__ __align__(16) _Float16 ldsP[4][16 * 40]; // per-wave P re-layout

  const int tid = threadIdx.x, lane = LANE, wid = WID;
  const int bh = blockIdx.y;          // b*8 + h
  const int b  = bh >> 3, h = bh & 7;
  const int q0 = blockIdx.x * 64 + wid * 16;
  const size_t qkv_base = (size_t)bh * 2048 * 64;

  // Q fragments live in registers for the whole kernel (DK=64 -> 2 chunks).
  const v16h aQ0 = load_frag_a(Qf + qkv_base + (size_t)q0 * 64,      64);
  const v16h aQ1 = load_frag_a(Qf + qkv_base + (size_t)q0 * 64 + 32, 64);

  v8f O[4] = {};
  float mrun[8], lrun[8];
#pragma unroll
  for (int r = 0; r < 8; ++r) { mrun[r] = -1e30f; lrun[r] = 0.f; }

  const int mrow0 = q0 + ((lane >> 4) << 3);              // + r = this lane's rows
  const float* distrow = dist + (size_t)b * 2048 * 2048;  // + m*2048 + key

  for (int kv = 0; kv < 2048; kv += 32) {
    // stage K chunk: ldsK[key][dk], contiguous dk (B-frags for Q@K^T)
    {
      const int key = tid >> 2, dko = (tid & 3) * 16;
      const v8h* src = (const v8h*)(Kf + qkv_base + (size_t)(kv + key) * 64 + dko);
      *(v8h*)(ldsK + key * 80 + dko)     = src[0];
      *(v8h*)(ldsK + key * 80 + dko + 8) = src[1];
    }
    // stage V chunk transposed: ldsV[dk][key] (B-frags for P@V)
    {
      const int dk = tid >> 1, kb = (tid & 1) * 16;
#pragma unroll
      for (int j = 0; j < 16; ++j)
        ldsV[dk * 40 + kb + j] = Vf[qkv_base + (size_t)(kv + kb + j) * 64 + dk];
    }
    __syncthreads();

    // logits: 16 q-rows x 32 keys as two 16x16 WMMA tiles
    v8f plog[2];
#pragma unroll
    for (int g = 0; g < 2; ++g) {
      v16h bk0 = load_frag_b(ldsK + g * 16 * 80,      80);
      v16h bk1 = load_frag_b(ldsK + g * 16 * 80 + 32, 80);
      v8f lg = {};
      lg = __builtin_amdgcn_wmma_f32_16x16x32_f16(false, aQ0, false, bk0, (short)0, lg, false, false);
      lg = __builtin_amdgcn_wmma_f32_16x16x32_f16(false, aQ1, false, bk1, (short)0, lg, false, false);
      const int key = kv + g * 16 + (lane & 15);
      const unsigned char mb = mask[(b << 11) + key];
#pragma unroll
      for (int r = 0; r < 8; ++r) {
        float v = lg[r] * distrow[(size_t)(mrow0 + r) * 2048 + key];
        lg[r] = mb ? -1e30f : v;
      }
      plog[g] = lg;
    }

    // online softmax (row reductions across the 16 key-lanes of each half)
#pragma unroll
    for (int r = 0; r < 8; ++r) {
      float cm = fmaxf(plog[0][r], plog[1][r]);
#pragma unroll
      for (int off = 1; off < 16; off <<= 1) cm = fmaxf(cm, __shfl_xor(cm, off, 16));
      const float mn   = fmaxf(mrun[r], cm);
      const float corr = __expf(mrun[r] - mn);
      const float p0 = __expf(plog[0][r] - mn);
      const float p1 = __expf(plog[1][r] - mn);
      float rs = p0 + p1;
#pragma unroll
      for (int off = 1; off < 16; off <<= 1) rs += __shfl_xor(rs, off, 16);
      lrun[r] = lrun[r] * corr + rs;
      mrun[r] = mn;
#pragma unroll
      for (int t = 0; t < 4; ++t) O[t][r] *= corr;
      // stash P (f16) row-major so it can be re-read as an A fragment
      const int ro = (r + ((lane >> 4) << 3)) * 40 + (lane & 15);
      ldsP[wid][ro]      = (_Float16)p0;
      ldsP[wid][ro + 16] = (_Float16)p1;
    }
    // O += P(16x32) @ V(32x64)   (same-wave LDS write->read: DScnt waits)
    v16h aP = load_frag_a(ldsP[wid], 40);
#pragma unroll
    for (int t = 0; t < 4; ++t) {
      v16h bv = load_frag_b(ldsV + t * 16 * 40, 40);
      O[t] = __builtin_amdgcn_wmma_f32_16x16x32_f16(
          false, aP, false, bv, (short)0, O[t], false, false);
    }
    __syncthreads();
  }

  // normalize and store to (B,S,D) f16 for the Wo projection
  const size_t obase = (size_t)b * 2048 * 512;
#pragma unroll
  for (int t = 0; t < 4; ++t) {
#pragma unroll
    for (int r = 0; r < 8; ++r) {
      const int m = mrow0 + r;
      const float v = O[t][r] / lrun[r];
      outp[obase + (size_t)m * 512 + h * 64 + t * 16 + (lane & 15)] = (_Float16)v;
    }
  }
}

// ---------------------------------------------------------------------------
// LayerNorm over rows of 512 floats. One wave per row (16 elems / lane).
// ---------------------------------------------------------------------------
__global__ __launch_bounds__(256)
void layernorm_kernel(const float* __restrict__ x, const float* __restrict__ g,
                      const float* __restrict__ bb, _Float16* __restrict__ out16,
                      float* __restrict__ out32)
{
  const int lane = LANE;
  const int row = blockIdx.x * 8 + WID;
  const float* xr = x + (size_t)row * 512;
  float vals[16], s = 0.f, ss = 0.f;
#pragma unroll
  for (int i = 0; i < 16; ++i) {
    float v = xr[lane + i * 32];
    vals[i] = v; s += v; ss += v * v;
  }
#pragma unroll
  for (int off = 1; off < 32; off <<= 1) {
    s  += __shfl_xor(s,  off, 32);
    ss += __shfl_xor(ss, off, 32);
  }
  const float mu  = s * (1.f / 512.f);
  const float var = ss * (1.f / 512.f) - mu * mu;
  const float rstd = rsqrtf(var + 1e-5f);
#pragma unroll
  for (int i = 0; i < 16; ++i) {
    const int c = lane + i * 32;
    const float y = (vals[i] - mu) * rstd * g[c] + bb[c];
    if (out16) out16[(size_t)row * 512 + c] = (_Float16)y;
    if (out32) out32[(size_t)row * 512 + c] = y;
  }
}

// ---------------------------------------------------------------------------
extern "C" void kernel_launch(void* const* d_in, const int* in_sizes, int n_in,
                              void* d_out, int out_size, void* d_ws, size_t ws_size,
                              hipStream_t stream)
{
  const float*         x    = (const float*)d_in[0];
  const float*         dist = (const float*)d_in[1];
  const unsigned char* mask = (const unsigned char*)d_in[2];
  const float* Wq = (const float*)d_in[3];  const float* bq = (const float*)d_in[4];
  const float* Wk = (const float*)d_in[5];  const float* bk = (const float*)d_in[6];
  const float* Wv = (const float*)d_in[7];  const float* bv = (const float*)d_in[8];
  const float* Wo = (const float*)d_in[9];  const float* bo = (const float*)d_in[10];
  const float* g1 = (const float*)d_in[11]; const float* b1 = (const float*)d_in[12];
  const float* g2 = (const float*)d_in[13]; const float* b2 = (const float*)d_in[14];
  const float* W1 = (const float*)d_in[15]; const float* bf1 = (const float*)d_in[16];
  const float* W2 = (const float*)d_in[17]; const float* bf2 = (const float*)d_in[18];

  // Workspace layout (72 MB total, with lifetime-based reuse):
  char* ws = (char*)d_ws;
  const size_t MB = 1024 * 1024;
  _Float16* Qf     = (_Float16*)(ws + 0 * MB);   // 8 MB  (B,H,S,DK) f16
  _Float16* Kf     = (_Float16*)(ws + 8 * MB);   // 8 MB
  _Float16* Vf     = (_Float16*)(ws + 16 * MB);  // 8 MB
  _Float16* attn16 = (_Float16*)(ws + 24 * MB);  // 8 MB  (B,S,D) f16
  float*    h1     = (float*)   (ws + 32 * MB);  // 16 MB x + attn_out (pre-LN1)
  _Float16* xn1_16 = (_Float16*)(ws + 48 * MB);  // 8 MB  LN1 out, FFN1 input
  float*    xn1_32 = (float*)   (ws + 56 * MB);  // 16 MB LN1 out, FFN2 residual
  _Float16* a1     = (_Float16*)(ws + 0 * MB);   // 32 MB relu(xn1@W1) (reuses QKV+attn)
  float*    h2     = (float*)   (ws + 32 * MB);  // 16 MB xn1 + ffn     (reuses h1)

  const dim3 blk256(256);

  // Q/K/V projections (Q pre-scaled by 1/sqrt(64) = 0.125)
  gemm_wmma_kernel<1, float><<<dim3(64, 4), blk256, 0, stream>>>(
      x, Wq, bq, nullptr, Qf, 8192, 512, 512, 0.125f, 0);
  gemm_wmma_kernel<1, float><<<dim3(64, 4), blk256, 0, stream>>>(
      x, Wk, bk, nullptr, Kf, 8192, 512, 512, 1.0f, 0);
  gemm_wmma_kernel<1, float><<<dim3(64, 4), blk256, 0, stream>>>(
      x, Wv, bv, nullptr, Vf, 8192, 512, 512, 1.0f, 0);

  // Flash attention with distance bias + mask
  attn_kernel<<<dim3(32, 32), dim3(128), 0, stream>>>(Qf, Kf, Vf, dist, mask, attn16);

  // Output projection + residual -> h1 = x + attn@Wo + bo
  gemm_wmma_kernel<2, _Float16><<<dim3(64, 4), blk256, 0, stream>>>(
      attn16, Wo, bo, x, h1, 8192, 512, 512, 1.0f, 0);

  // LN1 -> xn1 (f16 for FFN GEMM, f32 for second residual)
  layernorm_kernel<<<dim3(1024), blk256, 0, stream>>>(h1, g1, b1, xn1_16, xn1_32);

  // FFN1: relu(xn1 @ W1 + bf1) -> a1 (f16)
  gemm_wmma_kernel<0, _Float16><<<dim3(64, 16), blk256, 0, stream>>>(
      xn1_16, W1, bf1, nullptr, a1, 8192, 2048, 512, 1.0f, 1);

  // FFN2: a1 @ W2 + bf2 + xn1 -> h2 (f32)
  gemm_wmma_kernel<2, _Float16><<<dim3(64, 4), blk256, 0, stream>>>(
      a1, W2, bf2, xn1_32, h2, 8192, 512, 2048, 1.0f, 0);

  // LN2 -> final output (f32)
  layernorm_kernel<<<dim3(1024), blk256, 0, stream>>>(h2, g2, b2, nullptr, (float*)d_out);
}
